// Block_29729763623192
// MI455X (gfx1250) — matrix operand order repaired and checked by reading, not compiled, not run on hardware
//
#include <hip/hip_runtime.h>
#include <hip/hip_bf16.h>
#include <math.h>

// ---------------------------------------------------------------------------
// Problem constants (from reference)
// ---------------------------------------------------------------------------
#define BN   8
#define DCH  48
#define HH   160
#define WW   160
#define HWC  25600          // H*W
#define NP   204800         // B*H*W

typedef __attribute__((ext_vector_type(16))) _Float16 v16h;
typedef __attribute__((ext_vector_type(8)))  _Float16 v8h;
typedef __attribute__((ext_vector_type(8)))  float    v8f;
typedef __attribute__((ext_vector_type(4)))  int      v4i;

__device__ __forceinline__ float actf(float v, int act) {
  if (act == 1) return v >= 0.f ? v : 0.05f * v;                          // lrelu
  if (act == 2) return 0.5f * v * (1.f + erff(v * 0.70710678118654752f)); // gelu
  return v;
}

__device__ __forceinline__ v16h cat8(v8h lo, v8h hi) {
  return __builtin_shufflevector(lo, hi, 0, 1, 2, 3, 4, 5, 6, 7,
                                 8, 9, 10, 11, 12, 13, 14, 15);
}

// ---------------------------------------------------------------------------
// Fused 1x1 conv as WMMA GEMM:  Y[co,n] = act(W[co,:]·X[:,n] + bias) (+ res)
// X viewed as (B, ci_total, HW) with channel window [ci_off, ci_off+Ci).
// Y written into (B, co_total, HW) at channel offset co_off  -> free concat.
// 128 pixels per block (8 waves x 16-pixel subtiles), K padded to mult of 32.
// Weights zero-padded into LDS once; all fragment reads are contiguous LDS.
// ---------------------------------------------------------------------------
__global__ __launch_bounds__(256) void gemm1x1_wmma(
    const float* __restrict__ X, const float* __restrict__ Wt,
    const float* __restrict__ bias, const float* __restrict__ Res,
    float* __restrict__ Y,
    int Ci, int ci_total, int ci_off,
    int Co, int co_total, int co_off,
    int res_ct, int res_off, int act) {
  __shared__ _Float16 sx[128 * 160];   // sx[p][k], k padded (stride 160)
  __shared__ _Float16 sw[48 * 160];    // sw[m][k], zero padded
  const int tid = threadIdx.x;
  const int g0  = blockIdx.x * 128;    // global pixel base (HWC % 128 == 0)
  const int b   = g0 / HWC;
  const int p0  = g0 % HWC;
  const int cipad = ((Ci + 31) >> 5) << 5;

  __builtin_prefetch(Wt, 0, 3);        // global_prefetch_b8 (weights -> L2)

  // Stage weights (f32 -> f16), zero pad both M (to 48) and K (to 160).
  for (int idx = tid; idx < 48 * 160; idx += 256) {
    int m = idx / 160, k = idx - m * 160;
    float v = (m < Co && k < Ci) ? Wt[(size_t)m * Ci + k] : 0.f;
    sw[idx] = (_Float16)v;
  }
  // Stage activation tile transposed (coalesced read, strided LDS write).
  for (int idx = tid; idx < cipad * 128; idx += 256) {
    int c = idx >> 7, p = idx & 127;
    float v = (c < Ci) ? X[(size_t)(b * ci_total + ci_off + c) * HWC + p0 + p] : 0.f;
    sx[p * 160 + c] = (_Float16)v;
  }
  __syncthreads();

  const int wave = tid >> 5, lane = tid & 31;
  const int kgrp = lane >> 4, lrow = lane & 15;
  const int pxl  = wave * 16 + lrow;          // output pixel column of this lane
  const int mtiles  = (Co + 15) >> 4;
  const int kchunks = cipad >> 5;

  for (int mt = 0; mt < mtiles; ++mt) {
    v8f acc = {0.f, 0.f, 0.f, 0.f, 0.f, 0.f, 0.f, 0.f};
    const _Float16* wrow = &sw[(mt * 16 + lrow) * 160];
    const _Float16* xrow = &sx[pxl * 160];
    for (int kc = 0; kc < kchunks; ++kc) {
      // A: rows are contiguous in LDS; two 8-half runs per lane.
      v8h alo = *(const v8h*)(wrow + kc * 32 + kgrp * 8);
      v8h ahi = *(const v8h*)(wrow + kc * 32 + kgrp * 8 + 16);
      // B: same two 8-half runs per lane as A.
      v8h blo = *(const v8h*)(xrow + kc * 32 + kgrp * 8);
      v8h bhi = *(const v8h*)(xrow + kc * 32 + kgrp * 8 + 16);
      acc = __builtin_amdgcn_wmma_f32_16x16x32_f16(
          false, cat8(alo, ahi), false, cat8(blo, bhi), (short)0, acc,
          false, false);
    }
    // D: lanes 0-15 -> M=j, lanes 16-31 -> M=8+j ; N = lane&15
    const int p = p0 + pxl;
    #pragma unroll
    for (int j = 0; j < 8; ++j) {
      int m  = kgrp * 8 + j;
      int co = mt * 16 + m;
      if (co < Co) {
        float v = acc[j];
        if (bias) v += bias[co];
        v = actf(v, act);
        if (Res) v += Res[(size_t)(b * res_ct + res_off + co) * HWC + p];
        Y[(size_t)(b * co_total + co_off + co) * HWC + p] = v;
      }
    }
  }
}

// ---------------------------------------------------------------------------
// Depthwise 3x3 (SAME) + bias + activation, branch-free boundary handling.
// ---------------------------------------------------------------------------
__global__ __launch_bounds__(256) void dwconv3x3(
    const float* __restrict__ X, const float* __restrict__ Wd,
    const float* __restrict__ bd, float* __restrict__ Y,
    int C, int ci_total, int ci_off, int co_total, int co_off, int act) {
  int idx = blockIdx.x * 256 + threadIdx.x;
  if (idx >= BN * C * HWC) return;
  int p = idx % HWC;
  int c = (idx / HWC) % C;
  int b = idx / (HWC * C);
  int h = p / WW, w = p % WW;
  const float* xp = X + (size_t)(b * ci_total + ci_off + c) * HWC;
  float s = bd[c];
  #pragma unroll
  for (int dy = -1; dy <= 1; ++dy) {
    int hh = h + dy;
    bool hok = (unsigned)hh < (unsigned)HH;
    int hc = hok ? hh : h;
    #pragma unroll
    for (int dx = -1; dx <= 1; ++dx) {
      int ww = w + dx;
      bool ok = hok && ((unsigned)ww < (unsigned)WW);
      int wc = ((unsigned)ww < (unsigned)WW) ? ww : w;
      float wt = ok ? Wd[c * 9 + (dy + 1) * 3 + (dx + 1)] : 0.f;
      s += wt * xp[hc * WW + wc];     // address always valid; weight zeroed OOB
    }
  }
  Y[(size_t)(b * co_total + co_off + c) * HWC + p] = actf(s, act);
}

// ---------------------------------------------------------------------------
// In-place L2 normalization of each (b, channel) row over HW pixels.
// ---------------------------------------------------------------------------
__global__ __launch_bounds__(256) void l2norm_rows(float* __restrict__ Q) {
  __shared__ float red[256];
  __shared__ float sscale;
  float* q = Q + (size_t)blockIdx.x * HWC;
  float s = 0.f;
  for (int n = threadIdx.x; n < HWC; n += 256) { float v = q[n]; s += v * v; }
  red[threadIdx.x] = s;
  __syncthreads();
  for (int st = 128; st > 0; st >>= 1) {
    if (threadIdx.x < st) red[threadIdx.x] += red[threadIdx.x + st];
    __syncthreads();
  }
  if (threadIdx.x == 0) sscale = 1.f / fmaxf(sqrtf(red[0]), 1e-12f);
  __syncthreads();
  float sc = sscale;
  for (int n = threadIdx.x; n < HWC; n += 256) q[n] *= sc;
}

// ---------------------------------------------------------------------------
// Channel attention scores: S[(b,hd,c),d] = sum_n q[c,n]*k[d,n]  (24x24, K=HW)
// ---------------------------------------------------------------------------
__global__ __launch_bounds__(256) void attn_scores(
    const float* __restrict__ Q, const float* __restrict__ K,
    float* __restrict__ S) {
  __shared__ float red[256];
  int x = blockIdx.x;
  int b = x / 48, hd = (x / 24) % 2, c = x % 24;
  const float* q = Q + (size_t)(b * 48 + hd * 24 + c) * HWC;
  const float* k = K + (size_t)(b * 48 + hd * 24) * HWC;
  float acc[24];
  #pragma unroll
  for (int d = 0; d < 24; ++d) acc[d] = 0.f;
  for (int n = threadIdx.x; n < HWC; n += 256) {
    float qv = q[n];
    #pragma unroll
    for (int d = 0; d < 24; ++d) acc[d] += qv * k[(size_t)d * HWC + n];
  }
  for (int d = 0; d < 24; ++d) {
    red[threadIdx.x] = acc[d];
    __syncthreads();
    for (int st = 128; st > 0; st >>= 1) {
      if (threadIdx.x < st) red[threadIdx.x] += red[threadIdx.x + st];
      __syncthreads();
    }
    if (threadIdx.x == 0) S[(size_t)x * 24 + d] = red[0];
    __syncthreads();
  }
}

// Softmax over d (24) with per-head temperature. 384 rows, one block.
__global__ void attn_softmax24(float* __restrict__ S, const float* __restrict__ temp) {
  int r = threadIdx.x;
  if (r >= 384) return;
  int hd = (r / 24) % 2;
  float t = temp[hd];
  float* s = S + (size_t)r * 24;
  float z[24], m = -1e30f;
  #pragma unroll
  for (int d = 0; d < 24; ++d) { z[d] = s[d] * t; m = fmaxf(m, z[d]); }
  float sum = 0.f;
  #pragma unroll
  for (int d = 0; d < 24; ++d) { z[d] = expf(z[d] - m); sum += z[d]; }
  float inv = 1.f / sum;
  #pragma unroll
  for (int d = 0; d < 24; ++d) s[d] = z[d] * inv;
}

// out[c,n] = sum_d attn[c,d]*v[d,n]; grid = (HW/256, B*2)
// The 24x24 attention matrix (576 contiguous f32) is staged into LDS with
// CDNA5 async global->LDS transfers when the toolchain exposes them.
__global__ __launch_bounds__(256) void attn_apply(
    const float* __restrict__ S, const float* __restrict__ Vv,
    float* __restrict__ O) {
  __shared__ float sat[576];
  int bh = blockIdx.y, b = bh >> 1, hd = bh & 1;
#if __has_builtin(__builtin_amdgcn_global_load_async_to_lds_b128) && \
    __has_builtin(__builtin_amdgcn_s_wait_asynccnt)
  if (threadIdx.x < 144) {             // 144 lanes x 16B = 576 floats
    const float* gp = S + (size_t)bh * 576 + threadIdx.x * 4;
    __builtin_amdgcn_global_load_async_to_lds_b128(
        (__attribute__((address_space(1))) v4i*)gp,
        (__attribute__((address_space(3))) v4i*)&sat[threadIdx.x * 4],
        0, 0);
  }
  __builtin_amdgcn_s_wait_asynccnt(0);
#else
  for (int i = threadIdx.x; i < 576; i += 256) sat[i] = S[(size_t)bh * 576 + i];
#endif
  __syncthreads();
  int p = blockIdx.x * 256 + threadIdx.x;
  const float* v = Vv + (size_t)(b * 48 + hd * 24) * HWC + p;
  float vv[24];
  #pragma unroll
  for (int d = 0; d < 24; ++d) vv[d] = v[(size_t)d * HWC];
  #pragma unroll
  for (int c = 0; c < 24; ++c) {
    float s = 0.f;
    #pragma unroll
    for (int d = 0; d < 24; ++d) s += sat[c * 24 + d] * vv[d];
    O[(size_t)(b * 48 + hd * 24 + c) * HWC + p] = s;
  }
}

// ---------------------------------------------------------------------------
// Row/Col axial attention, fully fused per (b, line, half).
//   mode 0 (row):  line = h,  j runs over w (stride 1)
//   mode 1 (col):  line = w,  j runs over h (stride W)
// scores S[w,v] = sum_c q[c,w] k[c,v]  (WMMA, K=48 padded to 64)
// softmax over v, then out[c,w] = sum_v V[c,v] P[w,v]  (WMMA, K=160)
// Out[c,j] = gamma*out + Xres[c,j]; block covers 80 of the 160 target rows.
// sq/sk stored K-contiguous so fragment reads are ds_load_b128 pairs.
// grid = (2, 160, B), block = 256 (8 waves).
// ---------------------------------------------------------------------------
__global__ __launch_bounds__(256) void rowcol_attn_wmma(
    int mode, const float* __restrict__ Q, const float* __restrict__ K,
    const float* __restrict__ Vv, const float* __restrict__ Xres,
    float* __restrict__ Out, int out_ct, int out_co,
    const float* __restrict__ gamma) {
  __shared__ _Float16 sq[80 * 64];     // sq[m][c], c padded to 64 (zeros)
  __shared__ _Float16 sk[160 * 64];    // sk[v][c]
  __shared__ _Float16 sp[80 * 160];    // softmaxed scores P[m][v]
  const int tid  = threadIdx.x;
  const int half = blockIdx.x;
  const int L    = blockIdx.y;
  const int b    = blockIdx.z;
  const int j0   = half * 80;
  const int strj = (mode == 0) ? 1 : WW;
  const int boff = (mode == 0) ? L * WW : L;
  const size_t base = (size_t)b * 48 * HWC;

  // Stage q (coalesced over target positions) transposed into [m][c].
  for (int idx = tid; idx < 64 * 80; idx += 256) {
    int c = idx / 80, m = idx - c * 80;
    float v = (c < 48)
        ? Q[base + (size_t)c * HWC + boff + (size_t)(j0 + m) * strj] : 0.f;
    sq[m * 64 + c] = (_Float16)v;
  }
  // Stage k transposed into [v][c].
  for (int idx = tid; idx < 64 * 160; idx += 256) {
    int c = idx / 160, j = idx - c * 160;
    float v = (c < 48)
        ? K[base + (size_t)c * HWC + boff + (size_t)j * strj] : 0.f;
    sk[j * 64 + c] = (_Float16)v;
  }
  __syncthreads();

  const int wave = tid >> 5, lane = tid & 31;
  const int kgrp = lane >> 4, lrow = lane & 15;

  // ---- scores: 5 M-tiles x 10 N-tiles, K = 48 (padded to 64) ----
  for (int t = wave; t < 50; t += 8) {
    int mt = t / 10, nt = t - mt * 10;
    v8f acc = {0.f, 0.f, 0.f, 0.f, 0.f, 0.f, 0.f, 0.f};
    const _Float16* qrow = &sq[(mt * 16 + lrow) * 64];
    const _Float16* krow = &sk[(nt * 16 + lrow) * 64];
    #pragma unroll
    for (int kc = 0; kc < 2; ++kc) {
      v8h alo = *(const v8h*)(qrow + kc * 32 + kgrp * 8);
      v8h ahi = *(const v8h*)(qrow + kc * 32 + kgrp * 8 + 16);
      v8h blo = *(const v8h*)(krow + kc * 32 + kgrp * 8);
      v8h bhi = *(const v8h*)(krow + kc * 32 + kgrp * 8 + 16);
      acc = __builtin_amdgcn_wmma_f32_16x16x32_f16(
          false, cat8(alo, ahi), false, cat8(blo, bhi), (short)0, acc,
          false, false);
    }
    #pragma unroll
    for (int j = 0; j < 8; ++j) {
      int m = kgrp * 8 + j;
      sp[(mt * 16 + m) * 160 + nt * 16 + lrow] = (_Float16)acc[j];
    }
  }
  __syncthreads();

  // ---- softmax along source dim for each of the 80 local rows ----
  if (tid < 80) {
    _Float16* row = sp + tid * 160;
    float m = -1e30f;
    for (int v = 0; v < 160; ++v) m = fmaxf(m, (float)row[v]);
    float sum = 0.f;
    for (int v = 0; v < 160; ++v) {
      float e = expf((float)row[v] - m);
      row[v] = (_Float16)e;
      sum += e;
    }
    float inv = 1.f / sum;
    for (int v = 0; v < 160; ++v)
      row[v] = (_Float16)((float)row[v] * inv);
  }
  __syncthreads();

  // ---- out[c,w] = sum_v V[c,v] P[w,v] : 3 M-tiles x 5 N-tiles, K=160 ----
  const float g = gamma[0];
  for (int t = wave; t < 15; t += 8) {
    int mt = t / 5, nt = t - mt * 5;
    v8f acc = {0.f, 0.f, 0.f, 0.f, 0.f, 0.f, 0.f, 0.f};
    const int cm = mt * 16 + lrow;     // A row = channel
    const int wl = nt * 16 + lrow;     // B col = local target row
    const float* vrow = Vv + base + (size_t)cm * HWC + boff;
    const _Float16* prow = &sp[wl * 160];
    #pragma unroll
    for (int kc = 0; kc < 5; ++kc) {
      v16h a;
      if (mode == 0) {                 // contiguous: compiler merges loads
        const float* pb = vrow + kc * 32 + kgrp * 8;
        #pragma unroll
        for (int i = 0; i < 8; ++i) {
          a[i]     = (_Float16)pb[i];
          a[i + 8] = (_Float16)pb[i + 16];
        }
      } else {
        #pragma unroll
        for (int i = 0; i < 8; ++i) {
          int v0 = kc * 32 + kgrp * 8 + i;
          a[i]     = (_Float16)vrow[(size_t)v0 * WW];
          a[i + 8] = (_Float16)vrow[(size_t)(v0 + 16) * WW];
        }
      }
      v8h blo = *(const v8h*)(prow + kc * 32 + kgrp * 8);
      v8h bhi = *(const v8h*)(prow + kc * 32 + kgrp * 8 + 16);
      acc = __builtin_amdgcn_wmma_f32_16x16x32_f16(
          false, a, false, cat8(blo, bhi), (short)0, acc, false, false);
    }
    #pragma unroll
    for (int j = 0; j < 8; ++j) {
      int c  = mt * 16 + kgrp * 8 + j;
      int jg = j0 + nt * 16 + lrow;
      size_t off = boff + (size_t)jg * strj;
      float r = Xres[base + (size_t)c * HWC + off];
      Out[(size_t)(b * out_ct + out_co + c) * HWC + off] = g * acc[j] + r;
    }
  }
}

// ---------------------------------------------------------------------------
// Per-pixel LayerNorm over 96 channels. grid = NP/256.
// ---------------------------------------------------------------------------
__global__ __launch_bounds__(256) void layernorm96(
    const float* __restrict__ F, const float* __restrict__ g,
    const float* __restrict__ be, float* __restrict__ O) {
  int pix = blockIdx.x * 256 + threadIdx.x;
  int b = pix / HWC, p = pix % HWC;
  const float* f = F + (size_t)b * 96 * HWC + p;
  float s = 0.f, s2 = 0.f;
  for (int c = 0; c < 96; ++c) {
    float v = f[(size_t)c * HWC];
    s += v; s2 += v * v;
  }
  float mu  = s * (1.f / 96.f);
  float var = s2 * (1.f / 96.f) - mu * mu;
  float inv = rsqrtf(var + 1e-5f);
  float* o = O + (size_t)b * 96 * HWC + p;
  for (int c = 0; c < 96; ++c) {
    float v = f[(size_t)c * HWC];
    o[(size_t)c * HWC] = (v - mu) * inv * g[c] + be[c];
  }
}

// ---------------------------------------------------------------------------
// Host orchestration
// ---------------------------------------------------------------------------
extern "C" void kernel_launch(void* const* d_in, const int* in_sizes, int n_in,
                              void* d_out, int out_size, void* d_ws, size_t ws_size,
                              hipStream_t stream) {
  auto W = [&](int i) { return (const float*)d_in[i]; };
  const float* x = W(0);

  float* ws = (float*)d_ws;
  float* buf0   = ws;                        // 48 ch (t0/skip, x1, fea2)
  float* buf1   = ws + (size_t)48  * NP;     // 48 ch scratch
  float* buf2   = ws + (size_t)96  * NP;     // 48 ch scratch
  float* buf3   = ws + (size_t)144 * NP;     // 48 ch scratch
  float* cat96  = ws + (size_t)192 * NP;     // 96 ch (cat in A; attn_out; LN out)
  float* fea    = ws + (size_t)288 * NP;     // 96 ch (out_a | out_b)
  float* cat144 = ws + (size_t)384 * NP;     // 144 ch (o1 | o2 | o3)
  float* scr    = ws + (size_t)528 * NP;     // 9216 floats (attn scores)

  const int GB = NP / 128;  // 1600 blocks for gemm
  auto G = [&](const float* X, const float* Wt, const float* bias,
               const float* Res, float* Y, int Ci, int cit, int cio,
               int Co, int cot, int coo, int rct, int rco, int act) {
    gemm1x1_wmma<<<GB, 256, 0, stream>>>(X, Wt, bias, Res, Y, Ci, cit, cio,
                                         Co, cot, coo, rct, rco, act);
  };
  auto DWk = [&](const float* X, const float* Wd, const float* bd, float* Y,
                 int C, int cit, int cio, int cot, int coo, int act) {
    int total = BN * C * HWC;
    dwconv3x3<<<(total + 255) / 256, 256, 0, stream>>>(X, Wd, bd, Y, C, cit,
                                                       cio, cot, coo, act);
  };

  // ================= block_a =================
  G(x,    W(1),  W(2),  nullptr, buf0, 48, 48, 0, 48, 48, 0, 0, 0, 0);   // t0 = conv(x)
  G(buf0, W(3),  W(4),  nullptr, cat96, 48, 48, 0, 24, 96, 0, 0, 0, 0);  // d1
  G(buf0, W(5),  nullptr, nullptr, buf1, 48, 48, 0, 48, 48, 0, 0, 0, 0); // c1r pw
  DWk(buf1, W(6),  W(7),  buf2, 48, 48, 0, 48, 0, 1);                    // r1 = lrelu(dw)
  G(buf2, W(8),  W(9),  nullptr, cat96, 48, 48, 0, 24, 96, 24, 0, 0, 0); // d2
  G(buf2, W(10), nullptr, nullptr, buf1, 48, 48, 0, 48, 48, 0, 0, 0, 0); // c2r pw
  DWk(buf1, W(11), W(12), buf3, 48, 48, 0, 48, 0, 1);                    // r2
  G(buf3, W(13), W(14), nullptr, cat96, 48, 48, 0, 24, 96, 48, 0, 0, 0); // d3
  G(buf3, W(15), nullptr, nullptr, buf1, 48, 48, 0, 48, 48, 0, 0, 0, 0); // c3r pw
  DWk(buf1, W(16), W(17), buf2, 48, 48, 0, 48, 0, 1);                    // r3
  G(buf2, W(18), nullptr, nullptr, buf1, 48, 48, 0, 24, 24, 0, 0, 0, 0); // c4 pw (24ch)
  DWk(buf1, W(19), W(20), cat96, 24, 24, 0, 96, 72, 1);                  // r4 -> cat96[72:96]
  G(cat96, W(21), W(22), buf0, fea, 96, 96, 0, 48, 96, 0, 48, 0, 1);     // out_a = lrelu(c5)+skip

  // ================= block_b =================
  G(x,    W(23), nullptr, nullptr, buf1, 48, 48, 0, 48, 48, 0, 0, 0, 0); // conv1 pw
  DWk(buf1, W(24), W(25), buf2, 48, 48, 0, 48, 0, 2);                    // gelu(bsconv)
  G(x,    W(28), W(29), nullptr, buf3, 48, 48, 0, 48, 48, 0, 0, 0, 0);   // conv0(x)
  G(buf2, W(26), W(27), buf3, buf0, 48, 48, 0, 48, 48, 0, 48, 0, 0);     // x1

  // --- channel self-attention ---
  G(buf0, W(32), nullptr, nullptr, buf1, 48, 48, 0, 48, 48, 0, 0, 0, 0); // q
  G(buf0, W(33), nullptr, nullptr, buf2, 48, 48, 0, 48, 48, 0, 0, 0, 0); // k
  G(buf0, W(34), nullptr, nullptr, buf3, 48, 48, 0, 48, 48, 0, 0, 0, 0); // v
  l2norm_rows<<<BN * 48, 256, 0, stream>>>(buf1);
  l2norm_rows<<<BN * 48, 256, 0, stream>>>(buf2);
  attn_scores<<<BN * 2 * 24, 256, 0, stream>>>(buf1, buf2, scr);
  attn_softmax24<<<1, 384, 0, stream>>>(scr, W(36));
  attn_apply<<<dim3(HWC / 256, BN * 2), 256, 0, stream>>>(scr, buf3, cat96);
  G(cat96, W(35), nullptr, nullptr, cat144, 48, 48, 0, 48, 144, 0, 0, 0, 0); // o1

  // --- row attention ---
  G(buf0,   W(37), nullptr, nullptr, buf1, 48, 48, 0, 48, 48, 0, 0, 0, 0);   // q
  G(buf0,   W(38), nullptr, nullptr, buf2, 48, 48, 0, 48, 48, 0, 0, 0, 0);   // k
  G(cat144, W(39), nullptr, nullptr, buf3, 48, 144, 0, 48, 48, 0, 0, 0, 0);  // v from o1
  rowcol_attn_wmma<<<dim3(2, HH, BN), 256, 0, stream>>>(
      0, buf1, buf2, buf3, buf0, cat144, 144, 48, W(40));                    // o2

  // --- col attention ---
  G(buf0,   W(41), nullptr, nullptr, buf1, 48, 48, 0, 48, 48, 0, 0, 0, 0);
  G(buf0,   W(42), nullptr, nullptr, buf2, 48, 48, 0, 48, 48, 0, 0, 0, 0);
  G(cat144, W(43), nullptr, nullptr, buf3, 48, 144, 0, 48, 48, 0, 0, 0, 0);
  rowcol_attn_wmma<<<dim3(2, WW, BN), 256, 0, stream>>>(
      1, buf1, buf2, buf3, buf0, cat144, 144, 96, W(44));                    // o3

  G(cat144, W(30), W(31), nullptr, fea, 144, 144, 0, 48, 96, 48, 0, 0, 2);   // out_b

  // ================= blk =================
  layernorm96<<<NP / 256, 256, 0, stream>>>(fea, W(45), W(46), cat96);
  G(cat96, W(47), W(48), nullptr, buf0, 96, 96, 0, 48, 48, 0, 0, 0, 0);      // fea2
  G(buf0,  W(49), nullptr, nullptr, buf1, 24, 48, 0, 24, 24, 0, 0, 0, 0);    // conv1 pw
  DWk(buf1, W(50), W(51), buf2, 24, 24, 0, 48, 0, 2);                        // f1
  G(buf0,  W(52), W(53), nullptr, buf2, 24, 48, 24, 24, 48, 24, 0, 0, 2);    // f2
  G(buf2,  W(54), W(55), buf0, (float*)d_out, 48, 48, 0, 48, 48, 0, 48, 0, 2); // out
}
